// CrystalAttention_69810398429668
// MI455X (gfx1250) — compile-verified, matrix operand order checked by reading, not run on hardware
//
#include <hip/hip_runtime.h>

// ---------------------------------------------------------------------------
// CrystalAttention for MI455X (gfx1250, wave32, WMMA bf16)
//   D = 512, N = 4096, B*T = 8192
// Flash-style fused distance-softmax-attention, bf16 WMMA, f32 accum.
// ---------------------------------------------------------------------------

typedef __attribute__((ext_vector_type(16))) __bf16 v16bf;
typedef __attribute__((ext_vector_type(8)))  float  v8f;

#define EMBED_DIM 512
#define NUM_NEURONS 4096
#define MTOT 8192   // B*T

// Load a 16x32 bf16 WMMA operand tile (A layout; B uses the same lane layout
// with its source storing "columns as rows"). Per-lane: two 16B contiguous
// global_load_b128 of K-pairs.
//   element [m][k]  ->  src[(row0 + m) * ld + k0 + k]
__device__ __forceinline__ v16bf load_tile(const __bf16* __restrict__ src,
                                           int row0, int ld, int k0, int lane) {
    const int m  = lane & 15;
    const int hi = (lane >> 4) & 1;
    const __bf16* p = src + (size_t)(row0 + m) * ld + k0 + hi * 8;
    union { v16bf v; uint4 q[2]; } u;
    u.q[0] = *reinterpret_cast<const uint4*>(p);       // K = k0+hi*8      .. +8
    u.q[1] = *reinterpret_cast<const uint4*>(p + 16);  // K = k0+16+hi*8   .. +8
    return u.v;
}

__device__ __forceinline__ v8f wmma_bf16(v16bf a, v16bf b, v8f c) {
    return __builtin_amdgcn_wmma_f32_16x16x32_bf16(false, a, false, b,
                                                   (short)0, c, false, false);
}

// --------------------------- prep kernels ----------------------------------

// Row-wise: f32 -> bf16 copy + squared-norm per row (512 cols).
// Source is read exactly once -> non-temporal loads keep L2 for bf16 operands.
__global__ void __launch_bounds__(256)
prep_rows(const float* __restrict__ src, __bf16* __restrict__ dst,
          float* __restrict__ sq) {
    const int r = blockIdx.x, t = threadIdx.x;
    const float* s = src + (size_t)r * EMBED_DIM;
    __bf16* d = dst + (size_t)r * EMBED_DIM;
    float acc = 0.f;
    for (int i = t; i < EMBED_DIM; i += 256) {
        float v = __builtin_nontemporal_load(&s[i]);
        acc += v * v;
        d[i] = (__bf16)v;
    }
    __shared__ float red[256];
    red[t] = acc; __syncthreads();
    for (int s2 = 128; s2 > 0; s2 >>= 1) {
        if (t < s2) red[t] += red[t + s2];
        __syncthreads();
    }
    if (t == 0) sq[r] = red[0];
}

// w_o -> bf16 (row major), w_v -> bf16 transposed: wvT[k][j] = w_v[j][k]
__global__ void __launch_bounds__(256)
prep_w(const float* __restrict__ w_v, const float* __restrict__ w_o,
       __bf16* __restrict__ wvT, __bf16* __restrict__ woB) {
    const int i = blockIdx.x * 256 + threadIdx.x;      // < 512*512
    const int k = i >> 9, j = i & 511;
    woB[i] = (__bf16)w_o[i];
    wvT[i] = (__bf16)w_v[j * EMBED_DIM + k];
}

// bc[e] = sum_j w_o[e][j] * b_v[j] + b_o[e]
__global__ void __launch_bounds__(256)
prep_bc(const float* __restrict__ w_o, const float* __restrict__ b_v,
        const float* __restrict__ b_o, float* __restrict__ bc) {
    const int e = blockIdx.x, t = threadIdx.x;
    const float* wr = w_o + (size_t)e * EMBED_DIM;
    float acc = 0.f;
    for (int j = t; j < EMBED_DIM; j += 256) acc += wr[j] * b_v[j];
    __shared__ float red[256];
    red[t] = acc; __syncthreads();
    for (int s2 = 128; s2 > 0; s2 >>= 1) {
        if (t < s2) red[t] += red[t + s2];
        __syncthreads();
    }
    if (t == 0) bc[e] = red[0] + b_o[e];
}

// Wc[e][k] = sum_j w_o[e][j] * w_v[j][k]    (512x512, output bf16)
// A source = woB (row e, contiguous j); B source = wvT (row k, contiguous j)
__global__ void __launch_bounds__(256)
gemm_wc(const __bf16* __restrict__ A, const __bf16* __restrict__ Bs,
        __bf16* __restrict__ Wc) {
    const int w    = (blockIdx.x * blockDim.x + threadIdx.x) >> 5;
    const int lane = threadIdx.x & 31;
    const int e0 = (w >> 5) * 16, c0 = (w & 31) * 16;
    v8f acc = {};
    #pragma unroll 4
    for (int k0 = 0; k0 < EMBED_DIM; k0 += 32) {
        v16bf a = load_tile(A,  e0, EMBED_DIM, k0, lane);
        v16bf b = load_tile(Bs, c0, EMBED_DIM, k0, lane);
        acc = wmma_bf16(a, b, acc);
    }
    const int hi = lane >> 4, col = lane & 15;
    #pragma unroll
    for (int kv = 0; kv < 8; ++kv)
        Wc[(size_t)(e0 + kv + hi * 8) * EMBED_DIM + c0 + col] = (__bf16)acc[kv];
}

// V2T[e][n] = sum_k Wc[e][k] * pos[n][k] + bc[e]   (512 x 4096, bf16 out)
__global__ void __launch_bounds__(256)
gemm_v2(const __bf16* __restrict__ Wc, const __bf16* __restrict__ posB,
        const float* __restrict__ bc, __bf16* __restrict__ V2T) {
    const int w    = (blockIdx.x * blockDim.x + threadIdx.x) >> 5;
    const int lane = threadIdx.x & 31;
    const int e0 = (w >> 8) * 16, n0 = (w & 255) * 16;
    v8f acc = {};
    #pragma unroll 4
    for (int k0 = 0; k0 < EMBED_DIM; k0 += 32) {
        v16bf a = load_tile(Wc,   e0, EMBED_DIM, k0, lane);
        v16bf b = load_tile(posB, n0, EMBED_DIM, k0, lane);
        acc = wmma_bf16(a, b, acc);
    }
    const int hi = lane >> 4, col = lane & 15;
    #pragma unroll
    for (int kv = 0; kv < 8; ++kv) {
        const int row = e0 + kv + hi * 8;
        V2T[(size_t)row * NUM_NEURONS + n0 + col] = (__bf16)(acc[kv] + bc[row]);
    }
}

// --------------------------- fused main kernel -----------------------------
// Grid: 256 blocks x 256 threads. Block = 32 token rows.
// 8 waves: mi = row half (16 rows), di = 0..3 is both the score-column tile
// (16 of the 64-neuron block) and the 128-wide output D slice.
__global__ void __launch_bounds__(256)
crystal_flash(const __bf16* __restrict__ xB,  const float* __restrict__ x2,
              const __bf16* __restrict__ posB, const float* __restrict__ p2,
              const float* __restrict__ iscale, const __bf16* __restrict__ V2T,
              float* __restrict__ out) {
    __shared__ float  Sbuf[32][68];   // interaction scores (32 rows x 64 + pad)
    __shared__ __bf16 Pbuf[32][72];   // softmax numerators, bf16 (+pad, 16B rows)

    const int tid  = threadIdx.x;
    const int lane = tid & 31;
    const int wv   = tid >> 5;
    const int mi   = wv & 1;
    const int di   = wv >> 1;
    const int hi   = lane >> 4;
    const int l15  = lane & 15;

    const int m0 = blockIdx.x * 32 + mi * 16;   // this wave's 16 token rows
    const int d0 = di * 128;                    // this wave's output cols

    // x2 for the 16 rows, pre-mapped to the C-tile (kv, hi) row layout
    float x2r[8];
    #pragma unroll
    for (int kv = 0; kv < 8; ++kv) x2r[kv] = x2[m0 + kv + hi * 8];

    v8f o[8] = {};                    // 16 x 128 f32 accumulator (64 VGPRs)
    float m_run = -__builtin_inff();
    float l_run = 0.f;
    const int row16 = mi * 16 + l15;  // row this lane owns in softmax phase

    for (int it = 0; it < NUM_NEURONS / 64; ++it) {
        const int n0 = it * 64;

        // prefetch next neuron block (pos rows + V2T row segments) toward WGP
        {
            const int n0p = (n0 + 64) & (NUM_NEURONS - 1);
            __builtin_prefetch(&posB[(size_t)(n0p + di * 16 + l15) * EMBED_DIM],
                               0, 3);
            __builtin_prefetch(&posB[(size_t)(n0p + di * 16 + l15) * EMBED_DIM
                                     + 256], 0, 3);
            #pragma unroll
            for (int t = 0; t < 8; ++t)
                __builtin_prefetch(&V2T[(size_t)(d0 + t * 16 + l15)
                                        * NUM_NEURONS + n0p], 0, 3);
        }

        // ---- S tile: x(16x512) @ pos^T -> 16x16 scores ----
        v8f s = {};
        #pragma unroll 4
        for (int k0 = 0; k0 < EMBED_DIM; k0 += 32) {
            v16bf a = load_tile(xB,   m0,           EMBED_DIM, k0, lane);
            v16bf b = load_tile(posB, n0 + di * 16, EMBED_DIM, k0, lane);
            s = wmma_bf16(a, b, s);
        }
        // ---- distance -> interaction, write to LDS ----
        const int ncol = n0 + di * 16 + l15;
        const float p2v = p2[ncol];
        const float scv = iscale[ncol];
        #pragma unroll
        for (int kv = 0; kv < 8; ++kv) {
            float sq = fmaxf(x2r[kv] - 2.f * s[kv] + p2v, 0.f);
            Sbuf[mi * 16 + kv + hi * 8][di * 16 + l15] =
                scv / (__builtin_sqrtf(sq) + 0.1f);
        }
        __syncthreads();

        // ---- online softmax (redundant per di; di==0 writes Pbuf) ----
        const int cb = hi * 32;       // lane covers 32 of the 64 cols
        float lm = -__builtin_inff();
        #pragma unroll 8
        for (int j = 0; j < 32; ++j) lm = fmaxf(lm, Sbuf[row16][cb + j]);
        lm = fmaxf(lm, __shfl_xor(lm, 16, 32));
        const float m_new = fmaxf(m_run, lm);
        const float alpha = __expf(m_run - m_new);
        float ps = 0.f;
        #pragma unroll 8
        for (int j = 0; j < 32; ++j) {
            float p = __expf(Sbuf[row16][cb + j] - m_new);
            ps += p;
            if (di == 0) Pbuf[row16][cb + j] = (__bf16)p;
        }
        ps += __shfl_xor(ps, 16, 32);
        l_run = l_run * alpha + ps;
        m_run = m_new;
        __syncthreads();

        // ---- rescale accumulator by per-row alpha ----
        float ak[8];
        #pragma unroll
        for (int kv = 0; kv < 8; ++kv) ak[kv] = __shfl(alpha, kv + hi * 8, 32);
        #pragma unroll
        for (int t = 0; t < 8; ++t)
            #pragma unroll
            for (int kv = 0; kv < 8; ++kv) o[t][kv] *= ak[kv];

        // ---- O += P(16x64) @ V2(64x128) ----
        #pragma unroll
        for (int kk = 0; kk < 64; kk += 32) {
            union { v16bf v; uint4 q[2]; } u;       // A operand from LDS
            const __bf16* pr = &Pbuf[mi * 16 + l15][kk + hi * 8];
            u.q[0] = *reinterpret_cast<const uint4*>(pr);
            u.q[1] = *reinterpret_cast<const uint4*>(pr + 16);
            #pragma unroll
            for (int t = 0; t < 8; ++t) {
                v16bf b = load_tile(V2T, d0 + t * 16, NUM_NEURONS, n0 + kk, lane);
                o[t] = wmma_bf16(u.v, b, o[t]);
            }
        }
        // next iteration's Pbuf writes are fenced by its post-Sbuf barrier
    }

    // ---- finalize: divide by row sums, non-temporal f32 store ----
    float li[8];
    #pragma unroll
    for (int kv = 0; kv < 8; ++kv)
        li[kv] = 1.f / __shfl(l_run, kv + hi * 8, 32);
    #pragma unroll
    for (int t = 0; t < 8; ++t)
        #pragma unroll
        for (int kv = 0; kv < 8; ++kv) {
            const int row = m0 + kv + hi * 8;
            __builtin_nontemporal_store(
                o[t][kv] * li[kv],
                &out[(size_t)row * EMBED_DIM + d0 + t * 16 + l15]);
        }
}

// --------------------------- launch ----------------------------------------

extern "C" void kernel_launch(void* const* d_in, const int* in_sizes, int n_in,
                              void* d_out, int out_size, void* d_ws, size_t ws_size,
                              hipStream_t stream) {
    (void)in_sizes; (void)n_in; (void)out_size; (void)ws_size;
    const float* x         = (const float*)d_in[0];
    const float* positions = (const float*)d_in[1];
    const float* iscale    = (const float*)d_in[2];
    const float* w_v       = (const float*)d_in[3];
    const float* b_v       = (const float*)d_in[4];
    const float* w_o       = (const float*)d_in[5];
    const float* b_o       = (const float*)d_in[6];
    float* out = (float*)d_out;

    char* ws = (char*)d_ws;                                   // ~18.4 MB used
    __bf16* xB   = (__bf16*)(ws);                             // 8 MB
    __bf16* posB = (__bf16*)(ws + (size_t)(8u  << 20));       // 4 MB
    __bf16* V2T  = (__bf16*)(ws + (size_t)(12u << 20));       // 4 MB
    __bf16* woB  = (__bf16*)(ws + (size_t)(16u << 20));       // 512 KB
    __bf16* wvT  = (__bf16*)(ws + (size_t)(16u << 20) + (512u  << 10));
    __bf16* Wc   = (__bf16*)(ws + (size_t)(16u << 20) + (1024u << 10));
    float*  x2   = (float*) (ws + (size_t)(16u << 20) + (1536u << 10));
    float*  p2   = (float*) (ws + (size_t)(16u << 20) + (1568u << 10));
    float*  bc   = (float*) (ws + (size_t)(16u << 20) + (1584u << 10));

    prep_rows<<<MTOT,        256, 0, stream>>>(x, xB, x2);
    prep_rows<<<NUM_NEURONS, 256, 0, stream>>>(positions, posB, p2);
    prep_w   <<<1024,        256, 0, stream>>>(w_v, w_o, wvT, woB);
    prep_bc  <<<EMBED_DIM,   256, 0, stream>>>(w_o, b_v, b_o, bc);
    gemm_wc  <<<128,         256, 0, stream>>>(woB, wvT, Wc);      // 512x512x512
    gemm_v2  <<<1024,        256, 0, stream>>>(Wc, posB, bc, V2T); // 512x4096x512
    crystal_flash<<<MTOT / 32, 256, 0, stream>>>(xB, x2, posB, p2, iscale,
                                                 V2T, out);
}
